// CrossAttentionBlock_28020366639331
// MI455X (gfx1250) — matrix operand order repaired and checked by reading, not compiled, not run on hardware
//
#include <hip/hip_runtime.h>
#include <hip/hip_bf16.h>

typedef __attribute__((ext_vector_type(16))) __bf16 v16bf;
typedef __attribute__((ext_vector_type(8)))  float  v8f;

#define SCALE_QK 0.17677669529663687f   // 1/sqrt(32)
#define NEGINF   (-1e30f)

// ---------------------------------------------------------------- utilities

__device__ __forceinline__ v8f wmma_bf16(v16bf a, v16bf b, v8f c) {
    // D = A(16x32) * B(32x16) + C, f32 accumulate
    return __builtin_amdgcn_wmma_f32_16x16x32_bf16(
        false, a, false, b, (short)0, c, false, false);
}

__device__ __forceinline__ v8f zero8() {
    v8f z;
#pragma unroll
    for (int i = 0; i < 8; ++i) z[i] = 0.0f;
    return z;
}

// A-fragment (16x32 bf16, row-major with leading dim ldk):
// lane l (m=l&15, kh=l>>4): VGPR0-3 <- A[m][kc+kh*16 .. +7], VGPR4-7 <- A[m][kc+kh*16+8 .. +7]
__device__ __forceinline__ v16bf load_afrag(const __bf16* base, int ldk, int m, int kc, int kh) {
    v16bf a;
    const __bf16* p0 = base + m * ldk + kc + kh * 16;
    const __bf16* p1 = p0 + 8;
    ((uint4*)&a)[0] = *(const uint4*)p0;
    ((uint4*)&a)[1] = *(const uint4*)p1;
    return a;
}

// B-fragment (32x16 bf16): lane l (n=l&15, kh=l>>4) needs B[kh*16+2i..2i+1][n] packed,
// i.e. 16 contiguous K values at fixed n when B is stored n-major (rows of K).
__device__ __forceinline__ v16bf load_bfrag(const __bf16* p) {
    v16bf b;
    ((uint4*)&b)[0] = *(const uint4*)p;
    ((uint4*)&b)[1] = *(const uint4*)(p + 8);
    return b;
}

// ---------------------------------------------------------------- prep kernels

// generic f32 -> (bf16 hi, bf16 lo) split
__global__ void __launch_bounds__(256) prep_split_kernel(
    const float* __restrict__ in, __bf16* __restrict__ oh, __bf16* __restrict__ ol, int n) {
    int i = blockIdx.x * 256 + threadIdx.x;
    if (i < n) {
        float x = in[i];
        __bf16 h = (__bf16)x;
        oh[i] = h;
        ol[i] = (__bf16)(x - (float)h);
    }
}

// conv_w [co][ci][3][3] -> wt[(dy*3+dx)][co][ci] split to bf16 hi/lo
__global__ void __launch_bounds__(256) prep_convw_kernel(
    const float* __restrict__ w, __bf16* __restrict__ oh, __bf16* __restrict__ ol) {
    int i = blockIdx.x * 256 + threadIdx.x;   // 9*65536 total
    int p  = i >> 16;
    int r  = i & 65535;
    int co = r >> 8;
    int ci = r & 255;
    float x = w[(co * 256 + ci) * 9 + p];
    __bf16 h = (__bf16)x;
    oh[i] = h;
    ol[i] = (__bf16)(x - (float)h);
}

// ---------------------------------------------------------------- shared phases

// LDS layout for proj kernels (33792 bytes):
//   sh  : __bf16[16*256]  @ 0      (normalized activations, hi)
//   sl  : __bf16[16*256]  @ 8192   (lo)
//   xc  : float [16*256]  @ 16384  (raw f32 rows)
//   red : float [256]     @ 32768  (reduction scratch)

__device__ void rmsnorm_split_lds(float* xc, float* red, __bf16* sh, __bf16* sl,
                                  const float* __restrict__ g, int tid) {
    int row = tid >> 4, seg = tid & 15;
    const float* p = xc + row * 256 + seg * 16;
    float s = 0.0f;
#pragma unroll
    for (int j = 0; j < 16; ++j) s += p[j] * p[j];
    red[row * 16 + seg] = s;
    __syncthreads();
    if (tid < 16) {
        float t = 0.0f;
#pragma unroll
        for (int j = 0; j < 16; ++j) t += red[tid * 16 + j];
        red[tid * 16] = rsqrtf(t * (1.0f / 256.0f) + 1e-6f);
    }
    __syncthreads();
    float gc = g[tid];
#pragma unroll
    for (int m = 0; m < 16; ++m) {
        float v = xc[m * 256 + tid] * red[m * 16] * gc;
        __bf16 h = (__bf16)v;
        sh[m * 256 + tid] = h;
        sl[m * 256 + tid] = (__bf16)(v - (float)h);
    }
    __syncthreads();
}

// 16x256 = (16x256 @ 256x256^T) + bias, store split bf16 in [h][row][hd] layout
__device__ void proj_gemm_store(const __bf16* sh, const __bf16* sl,
                                const __bf16* __restrict__ wh, const __bf16* __restrict__ wl,
                                const float* __restrict__ bias,
                                __bf16* __restrict__ oh, __bf16* __restrict__ ol,
                                int rows, int row0, int tid) {
    int wave = tid >> 5, lane = tid & 31, ln = lane & 15, kh = lane >> 4;
#pragma unroll
    for (int nt2 = 0; nt2 < 2; ++nt2) {
        int n0 = (wave * 2 + nt2) * 16;
        int co = n0 + ln;
        v8f acc = zero8();
        for (int kc = 0; kc < 256; kc += 32) {
            v16bf ah = load_afrag(sh, 256, ln, kc, kh);
            v16bf al = load_afrag(sl, 256, ln, kc, kh);
            v16bf bh = load_bfrag(wh + co * 256 + kc + kh * 16);
            v16bf bl = load_bfrag(wl + co * 256 + kc + kh * 16);
            acc = wmma_bf16(ah, bh, acc);
            acc = wmma_bf16(ah, bl, acc);
            acc = wmma_bf16(al, bh, acc);
        }
        float bv = bias[co];
        int hh = co >> 5, hd = co & 31;
#pragma unroll
        for (int r = 0; r < 8; ++r) {
            int M = r + 8 * kh;
            float v = acc[r] + bv;
            size_t addr = ((size_t)hh * rows + row0 + M) * 32 + hd;
            __bf16 x = (__bf16)v;
            oh[addr] = x;
            ol[addr] = (__bf16)(v - (float)x);
        }
    }
}

// ---------------------------------------------------------------- k-side: rmsnorm + kproj

__global__ void __launch_bounds__(256) kproj_kernel(
    const float* __restrict__ kin, const float* __restrict__ gk,
    const __bf16* __restrict__ wkh, const __bf16* __restrict__ wkl,
    const float* __restrict__ bk,
    __bf16* __restrict__ kph, __bf16* __restrict__ kpl) {
    extern __shared__ char smem[];
    __bf16* sh = (__bf16*)smem;
    __bf16* sl = (__bf16*)(smem + 8192);
    float*  xc = (float*)(smem + 16384);
    float*  red = (float*)(smem + 32768);
    int tid = threadIdx.x;
    int b = blockIdx.x >> 6, kt = blockIdx.x & 63, key0 = kt * 16;
    // gather transposed: xc[key_row m][channel tid]
    const float* src = kin + ((size_t)b * 256 + tid) * 1024 + key0;
#pragma unroll
    for (int m = 0; m < 16; ++m) xc[m * 256 + tid] = src[m];
    __syncthreads();
    rmsnorm_split_lds(xc, red, sh, sl, gk, tid);
    proj_gemm_store(sh, sl, wkh, wkl, bk,
                    kph + (size_t)b * 8 * 1024 * 32,
                    kpl + (size_t)b * 8 * 1024 * 32,
                    1024, key0, tid);
}

// ---------------------------------------------------------------- q-side: conv3x3 + rmsnorm + qproj

__global__ void __launch_bounds__(256) convq_kernel(
    const float* __restrict__ qin,
    const __bf16* __restrict__ wth, const __bf16* __restrict__ wtl,
    const float* __restrict__ gq,
    const __bf16* __restrict__ wqh, const __bf16* __restrict__ wql,
    const float* __restrict__ bq,
    __bf16* __restrict__ qph, __bf16* __restrict__ qpl) {
    extern __shared__ char smem[];
    __bf16* sh = (__bf16*)smem;
    __bf16* sl = (__bf16*)(smem + 8192);
    float*  xc = (float*)(smem + 16384);
    float*  red = (float*)(smem + 32768);
    int tid = threadIdx.x;
    int wave = tid >> 5, lane = tid & 31, ln = lane & 15, kh = lane >> 4;
    int wg = blockIdx.x;
    int b = wg >> 8, rem = wg & 255;
    int y = rem >> 2, x0 = (rem & 3) * 16;
    int q0 = y * 64 + x0;

    v8f acc[2];
    acc[0] = zero8(); acc[1] = zero8();

    for (int p = 0; p < 9; ++p) {
        int dy = p / 3 - 1, dx = p % 3 - 1;
        int yy = y + dy;
        // stage shifted input patch: rows = 16 x-positions, cols = 256 channels (tid = ci)
#pragma unroll
        for (int m = 0; m < 16; ++m) {
            int xx = x0 + m + dx;
            float v = 0.0f;
            if (yy >= 0 && yy < 64 && xx >= 0 && xx < 64)
                v = qin[(((size_t)b * 256 + tid) * 64 + yy) * 64 + xx];
            __bf16 h = (__bf16)v;
            sh[m * 256 + tid] = h;
            sl[m * 256 + tid] = (__bf16)(v - (float)h);
        }
        __syncthreads();
        const __bf16* bwh = wth + p * 65536;
        const __bf16* bwl = wtl + p * 65536;
        if (p < 8) __builtin_prefetch(wth + (p + 1) * 65536 + tid * 256, 0, 1);
#pragma unroll
        for (int nt2 = 0; nt2 < 2; ++nt2) {
            int co = (wave * 2 + nt2) * 16 + ln;
            for (int kc = 0; kc < 256; kc += 32) {
                v16bf ah = load_afrag(sh, 256, ln, kc, kh);
                v16bf al = load_afrag(sl, 256, ln, kc, kh);
                v16bf bh = load_bfrag(bwh + co * 256 + kc + kh * 16);
                v16bf bl = load_bfrag(bwl + co * 256 + kc + kh * 16);
                acc[nt2] = wmma_bf16(ah, bh, acc[nt2]);
                acc[nt2] = wmma_bf16(ah, bl, acc[nt2]);
                acc[nt2] = wmma_bf16(al, bh, acc[nt2]);
            }
        }
        __syncthreads();
    }
    // dump conv result into xc (f32)
#pragma unroll
    for (int nt2 = 0; nt2 < 2; ++nt2) {
        int co = (wave * 2 + nt2) * 16 + ln;
#pragma unroll
        for (int r = 0; r < 8; ++r)
            xc[(r + 8 * kh) * 256 + co] = acc[nt2][r];
    }
    __syncthreads();
    rmsnorm_split_lds(xc, red, sh, sl, gq, tid);
    proj_gemm_store(sh, sl, wqh, wql, bq,
                    qph + (size_t)b * 8 * 4096 * 32,
                    qpl + (size_t)b * 8 * 4096 * 32,
                    4096, q0, tid);
}

// ---------------------------------------------------------------- attention (scores+softmax+PV)

// LDS (110592 bytes): sc f32[8*16*192] @0 ; ath bf16[16*192] @98304 ; atl @104448
__global__ void __launch_bounds__(256) attn_kernel(
    const __bf16* __restrict__ qph, const __bf16* __restrict__ qpl,
    const __bf16* __restrict__ kph, const __bf16* __restrict__ kpl,
    const __bf16* __restrict__ vth, const __bf16* __restrict__ vtl,
    float* __restrict__ out) {
    extern __shared__ char smem[];
    float*  sc  = (float*)smem;
    __bf16* ath = (__bf16*)(smem + 98304);
    __bf16* atl = (__bf16*)(smem + 104448);
    int tid = threadIdx.x;
    int wave = tid >> 5, lane = tid & 31, ln = lane & 15, kh = lane >> 4;
    int b = blockIdx.x >> 8, qt = blockIdx.x & 255;
    int q0 = qt * 16, y = q0 >> 6, x0 = q0 & 63;

    // window of unmasked key tiles for this query tile (<= 6 ki rows x <= 2 half-rows = 12)
    float ci = 0.5f * (float)y - 0.25f;
    int ki_lo = (int)ceilf(ci - 3.0f);  if (ki_lo < 0)  ki_lo = 0;
    int ki_hi = (int)floorf(ci + 3.0f); if (ki_hi > 31) ki_hi = 31;
    float cj0 = 0.5f * (float)x0 - 0.25f;
    int kj_lo = (int)ceilf(cj0 - 3.0f);         if (kj_lo < 0)  kj_lo = 0;
    int kj_hi = (int)floorf(cj0 + 10.5f);       if (kj_hi > 31) kj_hi = 31;
    int jt0 = kj_lo >> 4, jt1 = kj_hi >> 4;
    int njt = jt1 - jt0 + 1;
    int NT = (ki_hi - ki_lo + 1) * njt;          // <= 12 real tiles

    for (int i = tid; i < 8 * 16 * 192; i += 256) sc[i] = NEGINF;
    __syncthreads();

    // ---- scores: wave == head
    {
        const __bf16* qr_h = qph + (((size_t)b * 8 + wave) * 4096 + q0) * 32;
        const __bf16* qr_l = qpl + (((size_t)b * 8 + wave) * 4096 + q0) * 32;
        v16bf ah = load_afrag(qr_h, 32, ln, 0, kh);
        v16bf al = load_afrag(qr_l, 32, ln, 0, kh);
        const __bf16* kb_h = kph + ((size_t)b * 8 + wave) * 1024 * 32;
        const __bf16* kb_l = kpl + ((size_t)b * 8 + wave) * 1024 * 32;
        for (int t = 0; t < NT; ++t) {
            int ki = ki_lo + t / njt;
            int jt = jt0 + t % njt;
            int kb = ki * 32 + jt * 16;
            v16bf bh = load_bfrag(kb_h + (size_t)(kb + ln) * 32 + kh * 16);
            v16bf bl = load_bfrag(kb_l + (size_t)(kb + ln) * 32 + kh * 16);
            v8f s = zero8();
            s = wmma_bf16(ah, bh, s);
            s = wmma_bf16(ah, bl, s);
            s = wmma_bf16(al, bh, s);
            int kj = jt * 16 + ln;
#pragma unroll
            for (int r = 0; r < 8; ++r) {
                int M = r + 8 * kh;
                float cj = 0.5f * (float)(x0 + M) - 0.25f;
                float val = (fabsf(cj - (float)kj) <= 3.0f) ? s[r] * SCALE_QK : NEGINF;
                sc[((wave * 16 + M) * 192) + t * 16 + ln] = val;
            }
        }
    }
    __syncthreads();

    // ---- softmax per (head,row); tid == h*16+row
    if (tid < 128) {
        float* p = sc + tid * 192;
        float mx = NEGINF;
        for (int j = 0; j < 192; ++j) mx = fmaxf(mx, p[j]);
        float s = 0.0f;
        for (int j = 0; j < 192; ++j) { float e = __expf(p[j] - mx); p[j] = e; s += e; }
        float inv = 1.0f / s;
        for (int j = 0; j < 192; ++j) p[j] *= inv;
    }
    __syncthreads();

    // ---- average heads, split to bf16
    for (int i = tid; i < 16 * 192; i += 256) {
        int row = i / 192, col = i - row * 192;
        float v = 0.0f;
#pragma unroll
        for (int h = 0; h < 8; ++h) v += sc[(h * 16 + row) * 192 + col];
        v *= 0.125f;
        __bf16 hh = (__bf16)v;
        ath[i] = hh;
        atl[i] = (__bf16)(v - (float)hh);
    }
    __syncthreads();

    // ---- PV: attn(16x192) @ V(192 keys x 256 ch); wave owns 2 n-tiles of channels
    v8f o[2];
    o[0] = zero8(); o[1] = zero8();
    for (int c = 0; c < 6; ++c) {
        v16bf ah = load_afrag(ath, 192, ln, c * 32, kh);
        v16bf al = load_afrag(atl, 192, ln, c * 32, kh);
        int st = 2 * c + kh;                 // key-tile slot this lane-half covers
        int kb = 0;                          // dummy slots: attn==0, any valid addr
        if (st < NT) {
            int ki = ki_lo + st / njt;
            int jt = jt0 + st % njt;
            kb = ki * 32 + jt * 16;
        }
#pragma unroll
        for (int nt2 = 0; nt2 < 2; ++nt2) {
            int ch = (wave * 2 + nt2) * 16 + ln;
            const __bf16* vb_h = vth + ((size_t)b * 256 + ch) * 1024 + kb;
            const __bf16* vb_l = vtl + ((size_t)b * 256 + ch) * 1024 + kb;
            v16bf bh = load_bfrag(vb_h);
            v16bf bl = load_bfrag(vb_l);
            o[nt2] = wmma_bf16(ah, bh, o[nt2]);
            o[nt2] = wmma_bf16(ah, bl, o[nt2]);
            o[nt2] = wmma_bf16(al, bh, o[nt2]);
        }
    }
#pragma unroll
    for (int nt2 = 0; nt2 < 2; ++nt2) {
        int ch = (wave * 2 + nt2) * 16 + ln;
        float* op = out + ((size_t)b * 256 + ch) * 4096 + q0;
#pragma unroll
        for (int r = 0; r < 8; ++r) op[r + 8 * kh] = o[nt2][r];
    }
}

// ---------------------------------------------------------------- launch

extern "C" void kernel_launch(void* const* d_in, const int* in_sizes, int n_in,
                              void* d_out, int out_size, void* d_ws, size_t ws_size,
                              hipStream_t stream) {
    (void)in_sizes; (void)n_in; (void)out_size; (void)ws_size;
    const float* q  = (const float*)d_in[0];
    const float* k  = (const float*)d_in[1];
    const float* v  = (const float*)d_in[2];
    const float* cw = (const float*)d_in[3];
    const float* gq = (const float*)d_in[4];
    const float* gk = (const float*)d_in[5];
    const float* wq = (const float*)d_in[6];
    const float* bq = (const float*)d_in[7];
    const float* wk = (const float*)d_in[8];
    const float* bk = (const float*)d_in[9];
    float* out = (float*)d_out;

    char* ws = (char*)d_ws;
    size_t off = 0;
    auto carve = [&](size_t bytes) -> char* {
        char* p = ws + off;
        off += (bytes + 255) & ~(size_t)255;
        return p;
    };
    __bf16* wt_h = (__bf16*)carve((size_t)9 * 65536 * 2);
    __bf16* wt_l = (__bf16*)carve((size_t)9 * 65536 * 2);
    __bf16* wq_h = (__bf16*)carve((size_t)65536 * 2);
    __bf16* wq_l = (__bf16*)carve((size_t)65536 * 2);
    __bf16* wk_h = (__bf16*)carve((size_t)65536 * 2);
    __bf16* wk_l = (__bf16*)carve((size_t)65536 * 2);
    __bf16* vt_h = (__bf16*)carve((size_t)524288 * 2);
    __bf16* vt_l = (__bf16*)carve((size_t)524288 * 2);
    __bf16* qp_h = (__bf16*)carve((size_t)2 * 8 * 4096 * 32 * 2);
    __bf16* qp_l = (__bf16*)carve((size_t)2 * 8 * 4096 * 32 * 2);
    __bf16* kp_h = (__bf16*)carve((size_t)2 * 8 * 1024 * 32 * 2);
    __bf16* kp_l = (__bf16*)carve((size_t)2 * 8 * 1024 * 32 * 2);

    prep_convw_kernel<<<2304, 256, 0, stream>>>(cw, wt_h, wt_l);
    prep_split_kernel<<<256,  256, 0, stream>>>(wq, wq_h, wq_l, 65536);
    prep_split_kernel<<<256,  256, 0, stream>>>(wk, wk_h, wk_l, 65536);
    prep_split_kernel<<<2048, 256, 0, stream>>>(v,  vt_h, vt_l, 524288);

    kproj_kernel<<<128, 256, 33792, stream>>>(k, gk, wk_h, wk_l, bk, kp_h, kp_l);
    convq_kernel<<<512, 256, 33792, stream>>>(q, wt_h, wt_l, gq, wq_h, wq_l, bq, qp_h, qp_l);
    attn_kernel<<<512, 256, 110592, stream>>>(qp_h, qp_l, kp_h, kp_l, vt_h, vt_l, out);
}
